// SpatialAttention_30734785970848
// MI455X (gfx1250) — matrix operand order repaired and checked by reading, not compiled, not run on hardware
//
#include <hip/hip_runtime.h>

typedef __bf16 bf16_t;
typedef __attribute__((ext_vector_type(16))) __bf16 bf16x16;
typedef __attribute__((ext_vector_type(8)))  float  floatx8;

#define TILE_M 128
#define TILE_N 128
#define TILE_K 32
#define KP     40    // padded LDS row (bf16 elems): 80B stride -> conflict-free b128 reads
#define NBUF   4     // async pipeline depth (3 tiles of lookahead)
#define THREADS 256
#define BUF_BYTES (TILE_M * KP * 2)   // 10240 B per buffer (A and B each)

// ---- gfx1250 async global->LDS copy (16B per lane), tracked by ASYNCcnt ----
__device__ __forceinline__ unsigned lds_off(const void* p) {
  // generic pointer to LDS: low 32 bits are the wave-relative LDS byte address
  return (unsigned)(unsigned long long)p;
}
__device__ __forceinline__ void async_b128(unsigned lds, const void* gaddr) {
  asm volatile("global_load_async_to_lds_b128 %0, %1, off"
               :: "v"(lds), "v"((unsigned long long)gaddr) : "memory");
}
template <int N>
__device__ __forceinline__ void wait_async() {
  asm volatile("s_wait_asynccnt %0" :: "i"(N) : "memory");
}

// ---------------- f32 -> bf16 conversion (x4 vectorized) ----------------
__global__ void __launch_bounds__(256)
cvt_f32_bf16_x4(const float4* __restrict__ in, uint2* __restrict__ out, long n4) {
  long i = (long)blockIdx.x * blockDim.x + threadIdx.x;
  if (i < n4) {
    const float4 v = in[i];
    alignas(8) bf16_t o[4] = {(bf16_t)v.x, (bf16_t)v.y, (bf16_t)v.z, (bf16_t)v.w};
    out[i] = *(const uint2*)o;
  }
}

// ---------------- generic bf16 WMMA GEMM ----------------
// C[m,n] = sum_k A[m,k] * Bt[n,k]  (+ bias[n])
// A  : bf16 [M,K] row-major (lda), batched by aBatch elements
// Bt : bf16 [N,K] row-major (ldb), batched by bBatch elements
// C  : OUT_TRANS ? C[n*ldc + m] : C[m*ldc + n]; f32 or bf16
// 256 threads = 8 waves (2 over M x 4 over N); wave tile 64x32 = 8 WMMA accs.
// 4-deep LDS pipeline filled by global_load_async_to_lds_b128 (ASYNCcnt):
// tile i+3's DMA is issued while tile i computes. Steady-state loop is
// branch-free (always issue + s_wait_asynccnt 8); the last 4 tiles are peeled
// with static waits 8/4/0. Requires nIter % 4 == 0 and nIter >= 8 (holds:
// K is 1024 or 4096 -> nIter 32 or 128).
template <bool OUT_BF16, bool OUT_TRANS, bool HAS_BIAS>
__global__ void __launch_bounds__(THREADS)
gemm_bf16_wmma(const bf16_t* __restrict__ A, size_t aBatch, int lda,
               const bf16_t* __restrict__ Bt, size_t bBatch, int ldb,
               const float* __restrict__ bias,
               void* __restrict__ Cout, size_t cBatch, int ldc,
               int M, int N, int K) {
  __shared__ bf16_t As[NBUF][TILE_M * KP];
  __shared__ bf16_t Bs[NBUF][TILE_N * KP];

  const int tid   = threadIdx.x;
  const int lane  = tid & 31;
  const int wave  = tid >> 5;
  const int waveM = wave & 1;   // 2 waves over M
  const int waveN = wave >> 1;  // 4 waves over N
  const int half  = lane >> 4;
  const int l16   = lane & 15;

  const int mTile = blockIdx.y * TILE_M;
  const int nTile = blockIdx.x * TILE_N;
  const int batch = blockIdx.z;

  const bf16_t* Ab = A  + (size_t)batch * aBatch;
  const bf16_t* Bb = Bt + (size_t)batch * bBatch;

  floatx8 acc[4][2];
#pragma unroll
  for (int s = 0; s < 4; ++s)
#pragma unroll
    for (int t = 0; t < 2; ++t)
      acc[s][t] = (floatx8){0.f, 0.f, 0.f, 0.f, 0.f, 0.f, 0.f, 0.f};

  // staging coords: each thread DMAs 2 x b128 (16 bf16) for each of A and B
  const int idx0 = tid * 8;
  const int row0 = idx0 >> 5;   // / TILE_K
  const int col0 = idx0 & 31;

  const bf16_t* gA0 = Ab + (size_t)(mTile + row0)      * lda + col0;
  const bf16_t* gA1 = Ab + (size_t)(mTile + row0 + 64) * lda + col0;
  const bf16_t* gB0 = Bb + (size_t)(nTile + row0)      * ldb + col0;
  const bf16_t* gB1 = Bb + (size_t)(nTile + row0 + 64) * ldb + col0;

  const unsigned dA0 = lds_off(&As[0][row0 * KP + col0]);
  const unsigned dA1 = lds_off(&As[0][(row0 + 64) * KP + col0]);
  const unsigned dB0 = lds_off(&Bs[0][row0 * KP + col0]);
  const unsigned dB1 = lds_off(&Bs[0][(row0 + 64) * KP + col0]);

  const int nIter = K / TILE_K;

  auto issue = [&](int buf, int tile) {
    const size_t off = (size_t)tile * TILE_K;
    const unsigned b = (unsigned)buf * BUF_BYTES;
    async_b128(dA0 + b, gA0 + off);
    async_b128(dA1 + b, gA1 + off);
    async_b128(dB0 + b, gB0 + off);
    async_b128(dB1 + b, gB1 + off);
  };

  auto compute = [&](int buf) {
    // load ALL fragments first so the 12 ds_load_b128 issue back-to-back and
    // dscnt drains across the 8-WMMA burst
    const bf16_t* Bbase = &Bs[buf][(waveN * 32 + l16) * KP + 16 * half];
    const bf16_t* Abase = &As[buf][(waveM * 64 + l16) * KP];
    bf16x16 bfr[2], afr[4];
#pragma unroll
    for (int t = 0; t < 2; ++t) {     // B frag: lane(n=l16,half) K = 16h+0..15
      uint4 tmp[2];
      tmp[0] = *(const uint4*)(Bbase + t * 16 * KP);
      tmp[1] = *(const uint4*)(Bbase + t * 16 * KP + 8);
      __builtin_memcpy(&bfr[t], tmp, 32);
    }
#pragma unroll
    for (int s = 0; s < 4; ++s) {     // A frag: lane(m=l16,half) K = 16h+0..15
      const bf16_t* p = Abase + s * 16 * KP;
      uint4 tmp[2];
      tmp[0] = *(const uint4*)(p + 16 * half);
      tmp[1] = *(const uint4*)(p + 16 * half + 8);
      __builtin_memcpy(&afr[s], tmp, 32);
    }
#pragma unroll
    for (int s = 0; s < 4; ++s)
#pragma unroll
      for (int t = 0; t < 2; ++t)
        acc[s][t] = __builtin_amdgcn_wmma_f32_16x16x32_bf16(
            false, afr[s], false, bfr[t], (short)0, acc[s][t], false, false);
  };

  // prologue: DMA tiles 0..2 into buffers 0..2, then wait for group 0 only
  issue(0, 0);
  issue(1, 1);
  issue(2, 2);
  wait_async<8>();
  __syncthreads();

  // steady state: for all 4 sub-steps, i+3 < nIter -> branch-free body
  int it = 0;
  for (; it <= nIter - 8; it += NBUF) {
#pragma unroll
    for (int u = 0; u < NBUF; ++u) {      // buffer index == u (it % 4 == 0)
      issue((u + 3) & 3, it + u + 3);
      compute(u);
      wait_async<8>();   // only groups i+2, i+3 may remain -> group i+1 landed
      __syncthreads();
    }
  }

  // epilogue: last 4 tiles (it == nIter-4), statically scheduled waits
  issue(3, it + 3);          // tile nIter-1 into buffer 3
  compute(0);
  wait_async<8>();
  __syncthreads();
  compute(1);
  wait_async<4>();           // only group nIter-1 may remain
  __syncthreads();
  compute(2);
  wait_async<0>();           // everything landed
  __syncthreads();
  compute(3);                // no LDS consumers after this -> no barrier

  // ---- writeback: lane holds fixed n, 8 consecutive m (m0 .. m0+7) ----
#pragma unroll
  for (int s = 0; s < 4; ++s) {
#pragma unroll
    for (int t = 0; t < 2; ++t) {
      const int n  = nTile + waveN * 32 + t * 16 + l16;
      const int m0 = mTile + waveM * 64 + s * 16 + 8 * half;
      const float bv = HAS_BIAS ? bias[n] : 0.0f;
      alignas(16) float v[8];
#pragma unroll
      for (int r = 0; r < 8; ++r) v[r] = acc[s][t][r] + bv;

      if (OUT_TRANS) {
        if (OUT_BF16) {
          bf16_t* C = (bf16_t*)Cout + (size_t)batch * cBatch;
          alignas(16) bf16_t tmp[8];
#pragma unroll
          for (int r = 0; r < 8; ++r) tmp[r] = (bf16_t)v[r];
          *(uint4*)&C[(size_t)n * ldc + m0] = *(const uint4*)tmp;     // one b128
        } else {
          float* C = (float*)Cout + (size_t)batch * cBatch;
          *(float4*)&C[(size_t)n * ldc + m0]     = *(const float4*)&v[0];
          *(float4*)&C[(size_t)n * ldc + m0 + 4] = *(const float4*)&v[4];
        }
      } else {
        if (OUT_BF16) {
          bf16_t* C = (bf16_t*)Cout + (size_t)batch * cBatch;
#pragma unroll
          for (int r = 0; r < 8; ++r) C[(size_t)(m0 + r) * ldc + n] = (bf16_t)v[r];
        } else {
          float* C = (float*)Cout + (size_t)batch * cBatch;
#pragma unroll
          for (int r = 0; r < 8; ++r) C[(size_t)(m0 + r) * ldc + n] = v[r];
        }
      }
    }
  }
}

// ---------------- softmax over g for L_T[b][g][f] (g-major) ----------------
__global__ void __launch_bounds__(256)
softmax_cols(const float* __restrict__ Lt, bf16_t* __restrict__ Wt, int F) {
  const size_t base = (size_t)blockIdx.y * F * F;
  const int f = blockIdx.x * blockDim.x + threadIdx.x;
  const float* L = Lt + base;
  bf16_t* W = Wt + base;

  float m = -3.0e38f, s = 0.f;            // online softmax, pass 1
  for (int g = 0; g < F; ++g) {
    const float val = L[(size_t)g * F + f];
    const float mn = fmaxf(m, val);
    s = s * __expf(m - mn) + __expf(val - mn);
    m = mn;
  }
  const float inv = 1.0f / s;
  for (int g = 0; g < F; ++g) {           // pass 2: write bf16 weights (transposed)
    const float val = L[(size_t)g * F + f];
    W[(size_t)g * F + f] = (bf16_t)(__expf(val - m) * inv);
  }
}

// ---------------- host-side launch ----------------
extern "C" void kernel_launch(void* const* d_in, const int* in_sizes, int n_in,
                              void* d_out, int out_size, void* d_ws, size_t ws_size,
                              hipStream_t stream) {
  constexpr int B = 8, S = 4096, F = 1024, H = 1024;
  const float* x  = (const float*)d_in[0];
  const float* Wq = (const float*)d_in[1];
  const float* bq = (const float*)d_in[2];
  const float* Wk = (const float*)d_in[3];
  const float* bk = (const float*)d_in[4];
  const float* Wv = (const float*)d_in[5];
  const float* bv = (const float*)d_in[6];
  float* out = (float*)d_out;

  char* wsp = (char*)d_ws;
  auto take = [&](size_t bytes) -> char* {
    char* p = wsp;
    wsp += (bytes + 255) & ~(size_t)255;
    return p;
  };
  bf16_t* xb   = (bf16_t*)take((size_t)B * S * F * 2);   // x in bf16
  bf16_t* wqb  = (bf16_t*)take((size_t)F * F * 2);
  bf16_t* wkb  = (bf16_t*)take((size_t)F * F * 2);
  bf16_t* wvb  = (bf16_t*)take((size_t)H * F * 2);
  bf16_t* qT   = (bf16_t*)take((size_t)B * F * S * 2);   // q transposed [F,S]
  bf16_t* kT   = (bf16_t*)take((size_t)B * F * S * 2);   // k transposed [F,S]
  float*  LT   = (float*) take((size_t)B * F * F * 4);   // logits^T [g,f] f32
  bf16_t* WT   = (bf16_t*)take((size_t)B * F * F * 2);   // softmax^T [g,f] bf16
  bf16_t* attn = (bf16_t*)take((size_t)B * S * F * 2);   // attn_out [S,F] bf16

  {  // precision downconverts (one pass each, 4 elems/thread)
    long n4 = (long)B * S * F / 4;
    cvt_f32_bf16_x4<<<(unsigned)((n4 + 255) / 256), 256, 0, stream>>>(
        (const float4*)x, (uint2*)xb, n4);
    n4 = (long)F * F / 4;
    cvt_f32_bf16_x4<<<(unsigned)((n4 + 255) / 256), 256, 0, stream>>>(
        (const float4*)Wq, (uint2*)wqb, n4);
    cvt_f32_bf16_x4<<<(unsigned)((n4 + 255) / 256), 256, 0, stream>>>(
        (const float4*)Wk, (uint2*)wkb, n4);
    n4 = (long)H * F / 4;
    cvt_f32_bf16_x4<<<(unsigned)((n4 + 255) / 256), 256, 0, stream>>>(
        (const float4*)Wv, (uint2*)wvb, n4);
  }

  const dim3 blk(THREADS);

  {  // q = x Wq^T + bq -> qT ;  k = x Wk^T + bk -> kT   (M=S, N=F, K=F)
    dim3 grd(F / TILE_N, S / TILE_M, B);
    gemm_bf16_wmma<true, true, true><<<grd, blk, 0, stream>>>(
        xb, (size_t)S * F, F, wqb, 0, F, bq, qT, (size_t)F * S, S, S, F, F);
    gemm_bf16_wmma<true, true, true><<<grd, blk, 0, stream>>>(
        xb, (size_t)S * F, F, wkb, 0, F, bk, kT, (size_t)F * S, S, S, F, F);
  }
  {  // logits[f,g] = sum_s q[s,f] k[s,g] -> stored as LT[g,f]   (M=F, N=F, K=S)
    dim3 grd(F / TILE_N, F / TILE_M, B);
    gemm_bf16_wmma<false, true, false><<<grd, blk, 0, stream>>>(
        qT, (size_t)F * S, S, kT, (size_t)F * S, S, nullptr,
        LT, (size_t)F * F, F, F, F, S);
  }
  // softmax over g (axis -1 of logits) -> WT[g,f] bf16
  softmax_cols<<<dim3(F / 256, B), 256, 0, stream>>>(LT, WT, F);

  {  // attn[s,g] = sum_f x[s,f] * weights[f,g] ; Bt[n=g,k=f] = WT[g,f]
    dim3 grd(F / TILE_N, S / TILE_M, B);
    gemm_bf16_wmma<true, false, false><<<grd, blk, 0, stream>>>(
        xb, (size_t)S * F, F, WT, (size_t)F * F, F, nullptr,
        attn, (size_t)S * F, F, S, F, F);
  }
  {  // out[s,h] = sum_f attn[s,f] * Wv[h,f] + bv[h]  (f32 -> d_out)
    dim3 grd(H / TILE_N, S / TILE_M, B);
    gemm_bf16_wmma<false, false, true><<<grd, blk, 0, stream>>>(
        attn, (size_t)S * F, F, wvb, 0, F, bv,
        out, (size_t)S * H, H, S, H, F);
  }

  (void)in_sizes; (void)n_in; (void)out_size; (void)ws_size;
}
